// TopologicalAttention_39848706573732
// MI455X (gfx1250) — matrix operand order repaired and checked by reading, not compiled, not run on hardware
//
#include <hip/hip_runtime.h>

typedef __attribute__((ext_vector_type(16))) _Float16 v16h;
typedef __attribute__((ext_vector_type(8)))  _Float16 v8h;
typedef __attribute__((ext_vector_type(8)))  float    v8f;

#define B_   4
#define S_   2048
#define DM_  1024   // d_model == d_attn
#define DTOP 16

// ---------------------------------------------------------------------------
// WMMA fragment helpers (layouts per CDNA5 ISA 7.12.2, wave32)
// A 16x32 f16: lane L -> row M=L%16, K chunks [8c,8c+8) and [16+8c,16+8c+8), c=L/16
// B 32x16 f16 (stored as Bt[n][k], n-major): lane L -> col N=L%16, K chunks [8c,8c+8) and [16+8c,16+8c+8), c=L/16
// C/D 16x16 f32: lane L -> col N=L%16, VGPR v -> row M = v + 8*(L/16)
// ---------------------------------------------------------------------------
static __device__ __forceinline__ v16h load_a_frag(const _Float16* rowk0, int lane) {
    const _Float16* p = rowk0 + ((lane >> 4) << 3);     // + 0 or 8
    v8h lo = *(const v8h*)(p);
    v8h hi = *(const v8h*)(p + 16);
    return __builtin_shufflevector(lo, hi, 0,1,2,3,4,5,6,7,8,9,10,11,12,13,14,15);
}
static __device__ __forceinline__ v16h load_b_frag(const _Float16* colk0, int lane) {
    const _Float16* p = colk0 + ((lane >> 4) << 3);     // + 0 or 8
    v8h lo = *(const v8h*)(p);
    v8h hi = *(const v8h*)(p + 16);
    return __builtin_shufflevector(lo, hi, 0,1,2,3,4,5,6,7,8,9,10,11,12,13,14,15);
}
static __device__ __forceinline__ v8f wmma_f16(v16h a, v16h b, v8f c) {
    return __builtin_amdgcn_wmma_f32_16x16x32_f16(false, a, false, b, (short)0, c, false, false);
}

// ---------------------------------------------------------------------------
// Kernel 1: bias[s,t] = dot(top[s,t,:16], Wtop) + btop   (streams 268 MB once)
// ---------------------------------------------------------------------------
__global__ __launch_bounds__(256)
void topo_bias_kernel(const float* __restrict__ top, const float* __restrict__ Wtop,
                      const float* __restrict__ btop, float* __restrict__ bias) {
    size_t idx = (size_t)blockIdx.x * 256 + threadIdx.x;       // 0 .. S*S-1
    const float4* t4 = (const float4*)(top + idx * DTOP);
    float acc = btop[0];
    #pragma unroll
    for (int q = 0; q < 4; ++q) {
        float4 v = t4[q];
        acc += v.x * Wtop[4*q+0] + v.y * Wtop[4*q+1] + v.z * Wtop[4*q+2] + v.w * Wtop[4*q+3];
    }
    bias[idx] = acc;
}

// ---------------------------------------------------------------------------
// Kernel 2: C_f16[M=8192,N=1024] = A_f32 * W_f32 + bvec   (QKV projection)
// 128x128 tile / WG (256 thr = 8 waves in 4(M)x2(N)); K-step 32 through LDS f16
// ---------------------------------------------------------------------------
__global__ __launch_bounds__(256)
void qkv_gemm_kernel(const float* __restrict__ A, const float* __restrict__ W,
                     const float* __restrict__ bvec, _Float16* __restrict__ out) {
    __shared__ _Float16 sA[128][48];   // [m][k]  stride 96B (32B-aligned frags)
    __shared__ _Float16 sB[128][48];   // [n][k]  (W transposed at staging)
    const int tid  = threadIdx.x;
    const int lane = tid & 31, w = tid >> 5;
    const int wm = w & 3, wn = w >> 2;
    const int laneRow = lane & 15;
    const int m0 = blockIdx.y * 128, n0 = blockIdx.x * 128;

    v8f acc[2][4];
    #pragma unroll
    for (int i = 0; i < 2; ++i)
        #pragma unroll
        for (int j = 0; j < 4; ++j) acc[i][j] = (v8f){0.f,0.f,0.f,0.f,0.f,0.f,0.f,0.f};

    for (int k0 = 0; k0 < DM_; k0 += 32) {
        // stage A tile 128x32 f32 -> f16 (8 thr/row, 32 rows/pass, 4 passes)
        {
            int rt = tid >> 3, ct = (tid & 7) * 4;
            #pragma unroll
            for (int p = 0; p < 4; ++p) {
                int r = p * 32 + rt;
                float4 v = *(const float4*)(A + (size_t)(m0 + r) * DM_ + k0 + ct);
                sA[r][ct+0] = (_Float16)v.x; sA[r][ct+1] = (_Float16)v.y;
                sA[r][ct+2] = (_Float16)v.z; sA[r][ct+3] = (_Float16)v.w;
            }
        }
        // stage W tile 32x128 f32 -> f16 transposed (32 thr/row, 8 rows/pass)
        {
            int kt = tid >> 5, ct = (tid & 31) * 4;
            #pragma unroll
            for (int p = 0; p < 4; ++p) {
                int kk = p * 8 + kt;
                float4 v = *(const float4*)(W + (size_t)(k0 + kk) * DM_ + n0 + ct);
                sB[ct+0][kk] = (_Float16)v.x; sB[ct+1][kk] = (_Float16)v.y;
                sB[ct+2][kk] = (_Float16)v.z; sB[ct+3][kk] = (_Float16)v.w;
            }
        }
        __syncthreads();
        v16h af[2], bf[4];
        #pragma unroll
        for (int i = 0; i < 2; ++i) af[i] = load_a_frag(&sA[wm*32 + i*16 + laneRow][0], lane);
        #pragma unroll
        for (int j = 0; j < 4; ++j) bf[j] = load_b_frag(&sB[wn*64 + j*16 + laneRow][0], lane);
        #pragma unroll
        for (int i = 0; i < 2; ++i)
            #pragma unroll
            for (int j = 0; j < 4; ++j) acc[i][j] = wmma_f16(af[i], bf[j], acc[i][j]);
        __syncthreads();
    }
    const int lh = lane >> 4, ln = lane & 15;
    #pragma unroll
    for (int i = 0; i < 2; ++i)
        #pragma unroll
        for (int j = 0; j < 4; ++j) {
            int n = n0 + wn*64 + j*16 + ln;
            float bn = bvec[n];
            #pragma unroll
            for (int v = 0; v < 8; ++v) {
                int m = m0 + wm*32 + i*16 + lh*8 + v;
                out[(size_t)m * DM_ + n] = (_Float16)(acc[i][j][v] + bn);
            }
        }
}

// ---------------------------------------------------------------------------
// Kernel 3: scores[b,s,t] = (Q[b,s,:] . K[b,t,:]) * scale + bias[s,t]
// K is stored row-major [t][d] == exactly the Bt[n][k] WMMA layout -> fragments
// load straight from global (Q/K f16 working set is L2-resident).
// ---------------------------------------------------------------------------
__global__ __launch_bounds__(256)
void scores_gemm_kernel(const _Float16* __restrict__ Q, const _Float16* __restrict__ K,
                        const float* __restrict__ bias, float* __restrict__ scores) {
    const int z = blockIdx.z;
    const _Float16* Qb = Q + (size_t)z * S_ * DM_;
    const _Float16* Kb = K + (size_t)z * S_ * DM_;
    float* Sb = scores + (size_t)z * S_ * S_;
    const int tid = threadIdx.x, lane = tid & 31, w = tid >> 5;
    const int wm = w & 3, wn = w >> 2, laneRow = lane & 15;
    const int m0 = blockIdx.y * 128, n0 = blockIdx.x * 128;

    v8f acc[2][4];
    #pragma unroll
    for (int i = 0; i < 2; ++i)
        #pragma unroll
        for (int j = 0; j < 4; ++j) acc[i][j] = (v8f){0.f,0.f,0.f,0.f,0.f,0.f,0.f,0.f};

    const _Float16* aRow[2]; const _Float16* bRow[4];
    #pragma unroll
    for (int i = 0; i < 2; ++i) aRow[i] = Qb + (size_t)(m0 + wm*32 + i*16 + laneRow) * DM_;
    #pragma unroll
    for (int j = 0; j < 4; ++j) bRow[j] = Kb + (size_t)(n0 + wn*64 + j*16 + laneRow) * DM_;

    for (int k0 = 0; k0 < DM_; k0 += 32) {
        if (k0 + 64 < DM_) {                       // pull next tiles toward L0/L2
            __builtin_prefetch(aRow[0] + k0 + 64, 0, 0);
            __builtin_prefetch(bRow[0] + k0 + 64, 0, 0);
        }
        v16h af[2], bf[4];
        #pragma unroll
        for (int i = 0; i < 2; ++i) af[i] = load_a_frag(aRow[i] + k0, lane);
        #pragma unroll
        for (int j = 0; j < 4; ++j) bf[j] = load_b_frag(bRow[j] + k0, lane);
        #pragma unroll
        for (int i = 0; i < 2; ++i)
            #pragma unroll
            for (int j = 0; j < 4; ++j) acc[i][j] = wmma_f16(af[i], bf[j], acc[i][j]);
    }
    const float scale = 0.03125f;                  // 1/sqrt(1024)
    const int lh = lane >> 4, ln = lane & 15;
    #pragma unroll
    for (int i = 0; i < 2; ++i)
        #pragma unroll
        for (int j = 0; j < 4; ++j) {
            int n = n0 + wn*64 + j*16 + ln;
            #pragma unroll
            for (int v = 0; v < 8; ++v) {
                int m = m0 + wm*32 + i*16 + lh*8 + v;
                Sb[(size_t)m * S_ + n] = acc[i][j][v] * scale + bias[(size_t)m * S_ + n];
            }
        }
}

// ---------------------------------------------------------------------------
// Kernel 4: row softmax over 2048 keys; writes f16 attn IN PLACE over the f32
// scores row (attn row pitch = 4096 halves = 8KB = scores row pitch).
// ---------------------------------------------------------------------------
__global__ __launch_bounds__(256)
void softmax_kernel(const float* __restrict__ scores, _Float16* __restrict__ attn) {
    const size_t row = blockIdx.x;                 // 0 .. B*S-1
    const float* srow = scores + row * S_;
    _Float16* arow = attn + row * (size_t)(2 * S_);
    const int tid = threadIdx.x;
    __shared__ float red[256];

    float vals[8]; float mx = -3.402823466e38f;
    #pragma unroll
    for (int i = 0; i < 8; ++i) { vals[i] = srow[tid + i*256]; mx = fmaxf(mx, vals[i]); }
    red[tid] = mx; __syncthreads();
    for (int s = 128; s > 0; s >>= 1) { if (tid < s) red[tid] = fmaxf(red[tid], red[tid+s]); __syncthreads(); }
    mx = red[0]; __syncthreads();

    float e[8]; float sum = 0.f;
    #pragma unroll
    for (int i = 0; i < 8; ++i) { e[i] = __expf(vals[i] - mx); sum += e[i]; }
    red[tid] = sum; __syncthreads();
    for (int s = 128; s > 0; s >>= 1) { if (tid < s) red[tid] += red[tid+s]; __syncthreads(); }
    float inv = 1.f / red[0];
    #pragma unroll
    for (int i = 0; i < 8; ++i) arow[tid + i*256] = (_Float16)(e[i] * inv);
}

// ---------------------------------------------------------------------------
// Kernel 5: out[b,s,d] = sum_t attn[b,s,t] * V[b,t,d]
// attn (A) loads straight from global; V (B) needs K-major access -> stage
// 32x128 tile transposed through LDS.
// ---------------------------------------------------------------------------
__global__ __launch_bounds__(256)
void out_gemm_kernel(const _Float16* __restrict__ attn, const _Float16* __restrict__ V,
                     float* __restrict__ out) {
    __shared__ _Float16 sB[128][48];               // [d][t-slice]
    const int z = blockIdx.z;
    const _Float16* Ab = attn + (size_t)z * S_ * (2 * S_);   // pitch 4096 halves
    const _Float16* Vb = V + (size_t)z * S_ * DM_;
    float* Ob = out + (size_t)z * S_ * DM_;
    const int tid = threadIdx.x, lane = tid & 31, w = tid >> 5;
    const int wm = w & 3, wn = w >> 2, laneRow = lane & 15;
    const int m0 = blockIdx.y * 128, n0 = blockIdx.x * 128;

    v8f acc[2][4];
    #pragma unroll
    for (int i = 0; i < 2; ++i)
        #pragma unroll
        for (int j = 0; j < 4; ++j) acc[i][j] = (v8f){0.f,0.f,0.f,0.f,0.f,0.f,0.f,0.f};

    for (int k0 = 0; k0 < S_; k0 += 32) {
        // stage V[k0..k0+32][n0..n0+128] transposed (4 halves per thread per pass)
        {
            int kt = tid >> 5, ct = (tid & 31) * 4;
            #pragma unroll
            for (int p = 0; p < 4; ++p) {
                int kk = p * 8 + kt;
                ushort4 raw = *(const ushort4*)(Vb + (size_t)(k0 + kk) * DM_ + n0 + ct);
                const _Float16* h = (const _Float16*)&raw;
                sB[ct+0][kk] = h[0]; sB[ct+1][kk] = h[1];
                sB[ct+2][kk] = h[2]; sB[ct+3][kk] = h[3];
            }
        }
        __syncthreads();
        v16h af[2], bf[4];
        #pragma unroll
        for (int i = 0; i < 2; ++i)
            af[i] = load_a_frag(Ab + (size_t)(m0 + wm*32 + i*16 + laneRow) * (2*S_) + k0, lane);
        #pragma unroll
        for (int j = 0; j < 4; ++j) bf[j] = load_b_frag(&sB[wn*64 + j*16 + laneRow][0], lane);
        #pragma unroll
        for (int i = 0; i < 2; ++i)
            #pragma unroll
            for (int j = 0; j < 4; ++j) acc[i][j] = wmma_f16(af[i], bf[j], acc[i][j]);
        __syncthreads();
    }
    const int lh = lane >> 4, ln = lane & 15;
    #pragma unroll
    for (int i = 0; i < 2; ++i)
        #pragma unroll
        for (int j = 0; j < 4; ++j) {
            int n = n0 + wn*64 + j*16 + ln;
            #pragma unroll
            for (int v = 0; v < 8; ++v) {
                int m = m0 + wm*32 + i*16 + lh*8 + v;
                Ob[(size_t)m * DM_ + n] = acc[i][j][v];
            }
        }
}

// ---------------------------------------------------------------------------
extern "C" void kernel_launch(void* const* d_in, const int* in_sizes, int n_in,
                              void* d_out, int out_size, void* d_ws, size_t ws_size,
                              hipStream_t stream) {
    const float* input = (const float*)d_in[0];
    const float* top   = (const float*)d_in[3];
    const float* Wq    = (const float*)d_in[4];
    const float* bq    = (const float*)d_in[5];
    const float* Wk    = (const float*)d_in[6];
    const float* bk    = (const float*)d_in[7];
    const float* Wv    = (const float*)d_in[8];
    const float* bv    = (const float*)d_in[9];
    const float* Wtop  = (const float*)d_in[10];
    const float* btop  = (const float*)d_in[11];
    float* out = (float*)d_out;

    // workspace layout (128 MB total)
    char* ws = (char*)d_ws;
    _Float16* Qh    = (_Float16*)(ws);                          // 16 MB
    _Float16* Kh    = (_Float16*)(ws + (16ull << 20));          // 16 MB
    _Float16* Vh    = (_Float16*)(ws + (32ull << 20));          // 16 MB
    float*    biasM = (float*)   (ws + (48ull << 20));          // 16 MB
    float*    sc    = (float*)   (ws + (64ull << 20));          // 64 MB (f16 attn aliased)

    topo_bias_kernel<<<(S_ * S_) / 256, 256, 0, stream>>>(top, Wtop, btop, biasM);

    qkv_gemm_kernel<<<dim3(DM_/128, (B_*S_)/128), 256, 0, stream>>>(input, Wq, bq, Qh);
    qkv_gemm_kernel<<<dim3(DM_/128, (B_*S_)/128), 256, 0, stream>>>(input, Wk, bk, Kh);
    qkv_gemm_kernel<<<dim3(DM_/128, (B_*S_)/128), 256, 0, stream>>>(input, Wv, bv, Vh);

    scores_gemm_kernel<<<dim3(S_/128, S_/128, B_), 256, 0, stream>>>(Qh, Kh, biasM, sc);

    softmax_kernel<<<B_ * S_, 256, 0, stream>>>(sc, (_Float16*)sc);

    out_gemm_kernel<<<dim3(DM_/128, S_/128, B_), 256, 0, stream>>>((const _Float16*)sc, Vh, out);
}
